// VariableSelectionNetwork_35751307772463
// MI455X (gfx1250) — compile-verified, matrix operand order checked by reading, not run on hardware
//
#include <hip/hip_runtime.h>
#include <hip/hip_bf16.h>

// Sizes from the reference: B=64, T=512, V=32, D=64, H=64
#define S_TOT 32768
#define NV    32
#define DD    64
#define HH    64
#define VD    2048
#define EPSF  1e-5f

typedef _Float16 f16;
typedef __attribute__((ext_vector_type(16))) _Float16 v16h;
typedef __attribute__((ext_vector_type(8)))  _Float16 h8;
typedef __attribute__((ext_vector_type(8)))  float    v8f;
typedef __attribute__((ext_vector_type(4)))  float    f4v;

// ---------------- workspace layout (halfs) for pre-swizzled f16 weights ----
// Each WMMA B-fragment (32x16) is stored as 512 contiguous halfs:
//   dst[f*512 + chunk*256 + lane*8 + i]  <->  b[i + 8*chunk] of lane
#define E_W1   (2048 * 64)          // wW1
#define E_SK   (2048 * 32)          // wSkW
#define E_W2   (64 * 32)            // wW2
#define E_G    (32 * 32)            // wGW1 / wGW2
#define E_VM   (64 * 64)            // one per-variable matrix
#define OFF_W1  0
#define OFF_SK  (OFF_W1 + E_W1)
#define OFF_W2  (OFF_SK + E_SK)
#define OFF_G1  (OFF_W2 + E_W2)
#define OFF_G2  (OFF_G1 + E_G)
#define OFF_VW1 (OFF_G2 + E_G)
#define OFF_VW2 (OFF_VW1 + 32 * E_VM)
#define OFF_VG1 (OFF_VW2 + 32 * E_VM)
#define OFF_VG2 (OFF_VG1 + 32 * E_VM)
#define WS_HALF (OFF_VG2 + 32 * E_VM)   // 724992 halfs = ~1.42 MB

// D = A(16x32 f16) * B(32x16 f16) + C(16x16 f32)
__device__ __forceinline__ v8f wmma_f16(v16h a, v16h b, v8f c) {
    return __builtin_amdgcn_wmma_f32_16x16x32_f16(
        /*neg_a=*/false, a, /*neg_b=*/false, b,
        /*c_mod=*/(short)0, c, /*reuse_a=*/false, /*reuse_b=*/false);
}

// A-fragment from a global fp32 row pointer: 4x b128 loads + cvt.
__device__ __forceinline__ v16h a_from_g(const float* __restrict__ rowp, int kc, int half) {
    const f4v* p = (const f4v*)(rowp + kc + half * 8);
    f4v q0 = p[0], q1 = p[1], q2 = p[4], q3 = p[5];   // runs {kb..kb+7, kb+16..kb+23}
    v16h a;
#pragma unroll
    for (int i = 0; i < 4; ++i) {
        a[i]      = (f16)q0[i];
        a[i + 4]  = (f16)q1[i];
        a[i + 8]  = (f16)q2[i];
        a[i + 12] = (f16)q3[i];
    }
    return a;
}

// B-fragment from pre-swizzled f16 weights: two coalesced b128 loads, no cvt.
__device__ __forceinline__ v16h b_from_swz(const _Float16* __restrict__ Wf, int f, int lane) {
    const h8* p = (const h8*)(Wf + f * 512);
    h8 lo = p[lane];
    h8 hi = p[lane + 32];
    v16h b;
#pragma unroll
    for (int i = 0; i < 8; ++i) { b[i] = lo[i]; b[i + 8] = hi[i]; }
    return b;
}

// A-fragment from an LDS f16 row-major (16 x ld) buffer.
__device__ __forceinline__ v16h a_from_lds(const _Float16* base, int ld, int l16,
                                           int half, int kc) {
    const _Float16* p = base + l16 * ld + kc + half * 8;
    v16h a;
#pragma unroll
    for (int i = 0; i < 8; ++i) { a[i] = p[i]; a[i + 8] = p[16 + i]; }
    return a;
}

__device__ __forceinline__ float elu1f(float v) { return v > 0.f ? v : (__expf(v) - 1.f); }
__device__ __forceinline__ float sigmf(float v) { return 1.f / (1.f + __expf(-v)); }

// ---------------------------------------------------------------------------
// Kernel 0: swizzle all weights (fp32 row-major K x N) into fragment-ready
// f16 layout in d_ws. Tiny one-shot cost (~0.7M elements).
// ---------------------------------------------------------------------------
__device__ __forceinline__ void swz_one(const float* __restrict__ W, int N,
                                        _Float16* __restrict__ dst, int e) {
    const int i     = e & 7;
    const int lane  = (e >> 3) & 31;
    const int chunk = (e >> 8) & 1;
    const int f     = e >> 9;
    const int nt    = f % (N >> 4);
    const int kstep = f / (N >> 4);
    const int k     = kstep * 32 + (lane >> 4) * 8 + chunk * 16 + i;
    const int col   = nt * 16 + (lane & 15);
    dst[e] = (_Float16)W[k * N + col];
}

__global__ __launch_bounds__(256) void k0_swizzle(
    const float* __restrict__ wW1,  const float* __restrict__ wSkW,
    const float* __restrict__ wW2,  const float* __restrict__ wGW1,
    const float* __restrict__ wGW2, const float* __restrict__ vW1,
    const float* __restrict__ vW2,  const float* __restrict__ vG1,
    const float* __restrict__ vG2,  _Float16* __restrict__ ws)
{
    int idx = blockIdx.x * 256 + threadIdx.x;
    if (idx < OFF_SK)       { swz_one(wW1,  64, ws + OFF_W1, idx - OFF_W1); return; }
    if (idx < OFF_W2)       { swz_one(wSkW, 32, ws + OFF_SK, idx - OFF_SK); return; }
    if (idx < OFF_G1)       { swz_one(wW2,  32, ws + OFF_W2, idx - OFF_W2); return; }
    if (idx < OFF_G2)       { swz_one(wGW1, 32, ws + OFF_G1, idx - OFF_G1); return; }
    if (idx < OFF_VW1)      { swz_one(wGW2, 32, ws + OFF_G2, idx - OFF_G2); return; }
    if (idx < OFF_VW2) { int e = idx - OFF_VW1;
        swz_one(vW1 + (e / E_VM) * E_VM, 64, ws + OFF_VW1 + (e / E_VM) * E_VM, e % E_VM); return; }
    if (idx < OFF_VG1) { int e = idx - OFF_VW2;
        swz_one(vW2 + (e / E_VM) * E_VM, 64, ws + OFF_VW2 + (e / E_VM) * E_VM, e % E_VM); return; }
    if (idx < OFF_VG2) { int e = idx - OFF_VG1;
        swz_one(vG1 + (e / E_VM) * E_VM, 64, ws + OFF_VG1 + (e / E_VM) * E_VM, e % E_VM); return; }
    if (idx < WS_HALF) { int e = idx - OFF_VG2;
        swz_one(vG2 + (e / E_VM) * E_VM, 64, ws + OFF_VG2 + (e / E_VM) * E_VM, e % E_VM); return; }
}

// ---------------------------------------------------------------------------
// Kernel 1: weights GRN -> softmax weights (S, 32). 16 rows per block.
// 8 waves split K=2048 (256 each); ds_add_f32 reduction; wave 0 runs the tail.
// ---------------------------------------------------------------------------
__global__ __launch_bounds__(256) void k1_weights_grn(
    const float* __restrict__ x,    const _Float16* __restrict__ wsf,
    const float* __restrict__ wb1,  const float* __restrict__ wb2,
    const float* __restrict__ wGb1, const float* __restrict__ wGb2,
    const float* __restrict__ wSkb, const float* __restrict__ wLNg,
    const float* __restrict__ wLNb, float* __restrict__ wts_out)
{
    __shared__ float     hacc[16 * 64];
    __shared__ float     racc[16 * 32];
    __shared__ _Float16  hbuf[16 * 64];
    __shared__ _Float16  h2buf[16 * 32];

    const int tid  = threadIdx.x;
    const int wave = tid >> 5;
    const int lane = tid & 31;
    const int half = lane >> 4;
    const int l16  = lane & 15;
    const int s0   = blockIdx.x * 16;

    for (int i = tid; i < 16 * 64; i += 256) hacc[i] = 0.f;
    for (int i = tid; i < 16 * 32; i += 256) racc[i] = 0.f;
    __syncthreads();

    // ---- main GEMM: rows 16, K split 8 ways across waves ----
    v8f cW0 = {}, cW1 = {}, cW2 = {}, cW3 = {}, cS0 = {}, cS1 = {};
    const float* rowp = x + (size_t)(s0 + l16) * VD;
    const _Float16* W1f = wsf + OFF_W1;
    const _Float16* SKf = wsf + OFF_SK;
#pragma unroll 2
    for (int ks = 0; ks < 8; ++ks) {
        const int kstep = wave * 8 + ks;
        __builtin_prefetch(rowp + kstep * 32 + 32, 0, 1);   // global_prefetch_b8
        v16h a = a_from_g(rowp, kstep * 32, half);
        cW0 = wmma_f16(a, b_from_swz(W1f, kstep * 4 + 0, lane), cW0);
        cW1 = wmma_f16(a, b_from_swz(W1f, kstep * 4 + 1, lane), cW1);
        cW2 = wmma_f16(a, b_from_swz(W1f, kstep * 4 + 2, lane), cW2);
        cW3 = wmma_f16(a, b_from_swz(W1f, kstep * 4 + 3, lane), cW3);
        cS0 = wmma_f16(a, b_from_swz(SKf, kstep * 2 + 0, lane), cS0);
        cS1 = wmma_f16(a, b_from_swz(SKf, kstep * 2 + 1, lane), cS1);
    }
    // reduce K-partials (C/D layout: lane = col, vgpr r = row r + 8*half)
#pragma unroll
    for (int r = 0; r < 8; ++r) {
        const int M = half * 8 + r;
        atomicAdd(&hacc[M * 64 +  0 + l16], cW0[r]);
        atomicAdd(&hacc[M * 64 + 16 + l16], cW1[r]);
        atomicAdd(&hacc[M * 64 + 32 + l16], cW2[r]);
        atomicAdd(&hacc[M * 64 + 48 + l16], cW3[r]);
        atomicAdd(&racc[M * 32 +  0 + l16], cS0[r]);
        atomicAdd(&racc[M * 32 + 16 + l16], cS1[r]);
    }
    __syncthreads();

    if (wave == 0) {
        // h = elu(hacc + wb1) -> f16 scratch
        for (int i = lane; i < 16 * 64; i += 32)
            hbuf[i] = (f16)elu1f(hacc[i] + wb1[i & 63]);

        // h2 = h @ wW2 + wb2  (16x64 @ 64x32)
        const _Float16* W2f = wsf + OFF_W2;
        v8f c0 = {}, c1 = {};
#pragma unroll
        for (int kc = 0; kc < 64; kc += 32) {
            v16h a = a_from_lds(hbuf, 64, l16, half, kc);
            c0 = wmma_f16(a, b_from_swz(W2f, (kc >> 5) * 2 + 0, lane), c0);
            c1 = wmma_f16(a, b_from_swz(W2f, (kc >> 5) * 2 + 1, lane), c1);
        }
#pragma unroll
        for (int r = 0; r < 8; ++r) {
            const int M = half * 8 + r;
            h2buf[M * 32 +  0 + l16] = (f16)(c0[r] + wb2[ 0 + l16]);
            h2buf[M * 32 + 16 + l16] = (f16)(c1[r] + wb2[16 + l16]);
        }

        // gates: g1 = h2 @ wGW1, g2 = h2 @ wGW2  (K = 32)
        v8f g10 = {}, g11 = {}, g20 = {}, g21 = {};
        {
            v16h a = a_from_lds(h2buf, 32, l16, half, 0);
            g10 = wmma_f16(a, b_from_swz(wsf + OFF_G1, 0, lane), g10);
            g11 = wmma_f16(a, b_from_swz(wsf + OFF_G1, 1, lane), g11);
            g20 = wmma_f16(a, b_from_swz(wsf + OFF_G2, 0, lane), g20);
            g21 = wmma_f16(a, b_from_swz(wsf + OFF_G2, 1, lane), g21);
        }

        // gated + residual, LN(32), softmax(32), store
        float pre[2][8];
#pragma unroll
        for (int r = 0; r < 8; ++r) {
            const int M = half * 8 + r;
#pragma unroll
            for (int nt = 0; nt < 2; ++nt) {
                const int col = nt * 16 + l16;
                const float u1 = (nt ? g11[r] : g10[r]) + wGb1[col];
                const float u2 = (nt ? g21[r] : g20[r]) + wGb2[col];
                pre[nt][r] = sigmf(u1) * u2 + racc[M * 32 + col] + wSkb[col];
            }
        }
#pragma unroll
        for (int r = 0; r < 8; ++r) {
            float s = pre[0][r] + pre[1][r];
            float q = pre[0][r] * pre[0][r] + pre[1][r] * pre[1][r];
#pragma unroll
            for (int m = 1; m < 16; m <<= 1) {   // reduce within 16-lane half
                s += __shfl_xor(s, m, 32);
                q += __shfl_xor(q, m, 32);
            }
            const float mu   = s * (1.f / 32.f);
            const float var  = q * (1.f / 32.f) - mu * mu;
            const float rstd = rsqrtf(var + EPSF);
            const float n0 = (pre[0][r] - mu) * rstd * wLNg[ 0 + l16] + wLNb[ 0 + l16];
            const float n1 = (pre[1][r] - mu) * rstd * wLNg[16 + l16] + wLNb[16 + l16];
            float mx = fmaxf(n0, n1);
#pragma unroll
            for (int m = 1; m < 16; m <<= 1) mx = fmaxf(mx, __shfl_xor(mx, m, 32));
            const float e0 = __expf(n0 - mx), e1 = __expf(n1 - mx);
            float se = e0 + e1;
#pragma unroll
            for (int m = 1; m < 16; m <<= 1) se += __shfl_xor(se, m, 32);
            const float inv = 1.f / se;
            const int M = half * 8 + r;
            wts_out[(size_t)(s0 + M) * NV +  0 + l16] = e0 * inv;
            wts_out[(size_t)(s0 + M) * NV + 16 + l16] = e1 * inv;
        }
    }
}

// ---------------------------------------------------------------------------
// Kernel 2: 32 per-variable GRNs + weighted combine. 16 rows per block,
// each wave owns 4 variables (v = wave + 8*vi). Intermediates relayout
// through per-wave LDS f16 scratch; cross-wave reduce via ds_add_f32.
// ---------------------------------------------------------------------------
__global__ __launch_bounds__(256) void k2_var_grns(
    const float* __restrict__ x,    const _Float16* __restrict__ wsf,
    const float* __restrict__ vb1,  const float* __restrict__ vb2,
    const float* __restrict__ vGb1, const float* __restrict__ vGb2,
    const float* __restrict__ vLNg, const float* __restrict__ vLNb,
    const float* __restrict__ wts,  float* __restrict__ selected)
{
    __shared__ float    selacc[16 * 64];
    __shared__ _Float16 scr[8][16 * 64];   // 2 KB per wave

    const int tid  = threadIdx.x;
    const int wave = tid >> 5;
    const int lane = tid & 31;
    const int half = lane >> 4;
    const int l16  = lane & 15;
    const int s0   = blockIdx.x * 16;
    _Float16* sw = scr[wave];

    for (int i = tid; i < 16 * 64; i += 256) selacc[i] = 0.f;
    __syncthreads();

    v8f sa0 = {}, sa1 = {}, sa2 = {}, sa3 = {};   // selected accumulators

    for (int vi = 0; vi < 4; ++vi) {
        const int v = wave + vi * 8;
        const _Float16* W1f = wsf + OFF_VW1 + v * E_VM;
        const _Float16* W2f = wsf + OFF_VW2 + v * E_VM;
        const _Float16* G1f = wsf + OFF_VG1 + v * E_VM;
        const _Float16* G2f = wsf + OFF_VG2 + v * E_VM;

        // ---- mm1: hv = elu(x[:,v,:] @ W1 + b1) -> scratch ----
        {
            v8f c0 = {}, c1 = {}, c2 = {}, c3 = {};
            const float* rowp = x + (size_t)(s0 + l16) * VD + v * DD;
#pragma unroll
            for (int kc = 0; kc < 64; kc += 32) {
                v16h a = a_from_g(rowp, kc, half);
                const int fb = (kc >> 5) * 4;
                c0 = wmma_f16(a, b_from_swz(W1f, fb + 0, lane), c0);
                c1 = wmma_f16(a, b_from_swz(W1f, fb + 1, lane), c1);
                c2 = wmma_f16(a, b_from_swz(W1f, fb + 2, lane), c2);
                c3 = wmma_f16(a, b_from_swz(W1f, fb + 3, lane), c3);
            }
#pragma unroll
            for (int nt = 0; nt < 4; ++nt) {
                const int col = nt * 16 + l16;
                const float bia = vb1[v * HH + col];
                const v8f& c = nt == 0 ? c0 : nt == 1 ? c1 : nt == 2 ? c2 : c3;
#pragma unroll
                for (int r = 0; r < 8; ++r)
                    sw[(half * 8 + r) * 64 + col] = (f16)elu1f(c[r] + bia);
            }
        }

        // ---- mm2: hv2 = hv @ W2 + b2 -> scratch (read A first, then overwrite) ----
        {
            v16h a0 = a_from_lds(sw, 64, l16, half, 0);
            v16h a1 = a_from_lds(sw, 64, l16, half, 32);
            v8f c0 = {}, c1 = {}, c2 = {}, c3 = {};
#pragma unroll
            for (int kc = 0; kc < 64; kc += 32) {
                v16h a = kc ? a1 : a0;
                const int fb = (kc >> 5) * 4;
                c0 = wmma_f16(a, b_from_swz(W2f, fb + 0, lane), c0);
                c1 = wmma_f16(a, b_from_swz(W2f, fb + 1, lane), c1);
                c2 = wmma_f16(a, b_from_swz(W2f, fb + 2, lane), c2);
                c3 = wmma_f16(a, b_from_swz(W2f, fb + 3, lane), c3);
            }
#pragma unroll
            for (int nt = 0; nt < 4; ++nt) {
                const int col = nt * 16 + l16;
                const float bia = vb2[v * HH + col];
                const v8f& c = nt == 0 ? c0 : nt == 1 ? c1 : nt == 2 ? c2 : c3;
#pragma unroll
                for (int r = 0; r < 8; ++r)
                    sw[(half * 8 + r) * 64 + col] = (f16)(c[r] + bia);
            }
        }

        // ---- gates + skip, per tile; keep pre-LN values in registers ----
        float pre[4][8];
        {
            v16h a0 = a_from_lds(sw, 64, l16, half, 0);
            v16h a1 = a_from_lds(sw, 64, l16, half, 32);
#pragma unroll
            for (int nt = 0; nt < 4; ++nt) {
                const int col = nt * 16 + l16;
                v8f gc1 = {}, gc2 = {};
#pragma unroll
                for (int kc = 0; kc < 64; kc += 32) {
                    v16h a = kc ? a1 : a0;
                    const int f = (kc >> 5) * 4 + nt;
                    gc1 = wmma_f16(a, b_from_swz(G1f, f, lane), gc1);
                    gc2 = wmma_f16(a, b_from_swz(G2f, f, lane), gc2);
                }
                const float bg1 = vGb1[v * HH + col];
                const float bg2 = vGb2[v * HH + col];
#pragma unroll
                for (int r = 0; r < 8; ++r) {
                    const int M = half * 8 + r;
                    const float u1 = gc1[r] + bg1;
                    const float u2 = gc2[r] + bg2;
                    // skip add: fp32 x re-read (L2-hot)
                    pre[nt][r] = sigmf(u1) * u2 +
                                 x[(size_t)(s0 + M) * VD + v * DD + col];
                }
            }
        }

        // ---- LN over H=64 per row, scale by wts, accumulate ----
        float lng[4], lnb[4];
#pragma unroll
        for (int nt = 0; nt < 4; ++nt) {
            lng[nt] = vLNg[v * HH + nt * 16 + l16];
            lnb[nt] = vLNb[v * HH + nt * 16 + l16];
        }
#pragma unroll
        for (int r = 0; r < 8; ++r) {
            float s = pre[0][r] + pre[1][r] + pre[2][r] + pre[3][r];
            float q = pre[0][r] * pre[0][r] + pre[1][r] * pre[1][r] +
                      pre[2][r] * pre[2][r] + pre[3][r] * pre[3][r];
#pragma unroll
            for (int m = 1; m < 16; m <<= 1) {
                s += __shfl_xor(s, m, 32);
                q += __shfl_xor(q, m, 32);
            }
            const float mu   = s * (1.f / 64.f);
            const float var  = q * (1.f / 64.f) - mu * mu;
            const float rstd = rsqrtf(var + EPSF);
            const int   M    = half * 8 + r;
            const float wt   = wts[(size_t)(s0 + M) * NV + v];
            sa0[r] += ((pre[0][r] - mu) * rstd * lng[0] + lnb[0]) * wt;
            sa1[r] += ((pre[1][r] - mu) * rstd * lng[1] + lnb[1]) * wt;
            sa2[r] += ((pre[2][r] - mu) * rstd * lng[2] + lnb[2]) * wt;
            sa3[r] += ((pre[3][r] - mu) * rstd * lng[3] + lnb[3]) * wt;
        }
    }

    // cross-wave reduction of selected, then coalesced store
#pragma unroll
    for (int r = 0; r < 8; ++r) {
        const int M = half * 8 + r;
        atomicAdd(&selacc[M * 64 +  0 + l16], sa0[r]);
        atomicAdd(&selacc[M * 64 + 16 + l16], sa1[r]);
        atomicAdd(&selacc[M * 64 + 32 + l16], sa2[r]);
        atomicAdd(&selacc[M * 64 + 48 + l16], sa3[r]);
    }
    __syncthreads();
    for (int i = tid; i < 16 * 64; i += 256)
        selected[(size_t)s0 * HH + i] = selacc[i];
}

extern "C" void kernel_launch(void* const* d_in, const int* in_sizes, int n_in,
                              void* d_out, int out_size, void* d_ws, size_t ws_size,
                              hipStream_t stream) {
    (void)in_sizes; (void)n_in; (void)out_size; (void)ws_size;
    const float* x    = (const float*)d_in[0];
    const float* vW1  = (const float*)d_in[1];
    const float* vb1  = (const float*)d_in[2];
    const float* vW2  = (const float*)d_in[3];
    const float* vb2  = (const float*)d_in[4];
    const float* vGW1 = (const float*)d_in[5];
    const float* vGb1 = (const float*)d_in[6];
    const float* vGW2 = (const float*)d_in[7];
    const float* vGb2 = (const float*)d_in[8];
    const float* vLNg = (const float*)d_in[9];
    const float* vLNb = (const float*)d_in[10];
    const float* wW1  = (const float*)d_in[11];
    const float* wb1  = (const float*)d_in[12];
    const float* wW2  = (const float*)d_in[13];
    const float* wb2  = (const float*)d_in[14];
    const float* wGW1 = (const float*)d_in[15];
    const float* wGb1 = (const float*)d_in[16];
    const float* wGW2 = (const float*)d_in[17];
    const float* wGb2 = (const float*)d_in[18];
    const float* wSkW = (const float*)d_in[19];
    const float* wSkb = (const float*)d_in[20];
    const float* wLNg = (const float*)d_in[21];
    const float* wLNb = (const float*)d_in[22];

    float* selected = (float*)d_out;                       // (S, 64)
    float* weights  = (float*)d_out + (size_t)S_TOT * HH;  // (S, 32)
    _Float16* wsf   = (_Float16*)d_ws;                     // needs ~1.42 MB

    // 0) one-shot weight swizzle into fragment-ready f16 layout
    k0_swizzle<<<WS_HALF / 256, 256, 0, stream>>>(
        wW1, wSkW, wW2, wGW1, wGW2, vW1, vW2, vGW1, vGW2, wsf);

    dim3 grid(S_TOT / 16), block(256);
    k1_weights_grn<<<grid, block, 0, stream>>>(
        x, wsf, wb1, wb2, wGb1, wGb2, wSkb, wLNg, wLNb, weights);
    k2_var_grns<<<grid, block, 0, stream>>>(
        x, wsf, vb1, vb2, vGb1, vGb2, vLNg, vLNb, weights, selected);
}
